// EnhanceLayerLinear_31430570672438
// MI455X (gfx1250) — matrix operand order repaired and checked by reading, not compiled, not run on hardware
//
#include <hip/hip_runtime.h>

typedef __bf16 bf16;
typedef __attribute__((ext_vector_type(16))) __bf16 bf16x16;
typedef __attribute__((ext_vector_type(8)))  float   f32x8;
typedef __attribute__((ext_vector_type(4)))  unsigned int u32x4;
typedef __attribute__((ext_vector_type(2)))  unsigned int u32x2;

union Frag { bf16x16 bf; u32x4 q[2]; };

// LDS row strides (elements), padded for conflict-free b128 fragment reads:
//  - stage-1 tiles: 72 elems = 144 B  (36 banks -> 16 rows cover 64 banks)
//  - stage-2 tiles: 136 elems = 272 B (68 banks -> 16 rows cover 64 banks)
// Both are multiples of 16 B so ds_load_b128 stays aligned.
#define S1 72
#define S2 136

static __device__ __forceinline__ unsigned pack2(float a, float b) {
    union { bf16 h[2]; unsigned u; } t;
    t.h[0] = (bf16)a;
    t.h[1] = (bf16)b;
    return t.u;
}

static __device__ __forceinline__ u32x4 ld128(const bf16* p) {
    return *(const u32x4*)p;
}

// A fragment: 16x32 bf16. lane<16 (m=lane): V0-3=k0..k0+7, V4-7=k0+16..23
//             lane>=16 (m=lane-16):        V0-3=k0+8..15, V4-7=k0+24..31
static __device__ __forceinline__ void loadA(Frag& f, const bf16* base, int row0,
                                             int stride, int k0, int lane) {
    const int m = lane & 15, h = lane >> 4;
    const bf16* r = base + (row0 + m) * stride + k0 + h * 8;
    f.q[0] = ld128(r);
    f.q[1] = ld128(r + 16);
}

// B fragment: 32x16 bf16, B[k][n] = src[n][k] (src row-major [col][k]).
// lane<16 (n=lane): V0-3=k0..k0+7, V4-7=k0+16..23 ; lane>=16: V0-3=k0+8..15, V4-7=k0+24..31
static __device__ __forceinline__ void loadB(Frag& f, const bf16* base, int col0,
                                             int stride, int k0, int lane) {
    const int n = lane & 15, h = lane >> 4;
    const bf16* r = base + (col0 + n) * stride + k0 + h * 8;
    f.q[0] = ld128(r);
    f.q[1] = ld128(r + 16);
}

// Load one 64-wide K chunk of the x tile (256 rows) and W tile (128 rows)
// into LDS (fp32 -> packed bf16 on the fly).
static __device__ __forceinline__ void loadChunk(const float* __restrict__ x,
                                                 const float* __restrict__ W,
                                                 bf16* Xd, bf16* Wd,
                                                 int m0, int gbase, int k0, int t) {
#pragma unroll
    for (int i = 0; i < 16; ++i) {          // x: 256 rows x 64
        const int idx = t + i * 256;
        const int row = idx >> 4;
        const int c   = (idx & 15) * 4;
        const float4 v = *(const float4*)(x + (size_t)(m0 + row) * 4096 + k0 + c);
        u32x2 p = { pack2(v.x, v.y), pack2(v.z, v.w) };
        *(u32x2*)(Xd + row * S1 + c) = p;
    }
#pragma unroll
    for (int i = 0; i < 8; ++i) {           // W: 128 cols x 64
        const int idx = t + i * 256;
        const int row = idx >> 4;
        const int c   = (idx & 15) * 4;
        const float4 u = *(const float4*)(W + (size_t)(gbase + row) * 4096 + k0 + c);
        u32x2 q = { pack2(u.x, u.y), pack2(u.z, u.w) };
        *(u32x2*)(Wd + row * S1 + c) = q;
    }
}

// One 64-wide K chunk of WMMA work: 2 k-steps x 16 WMMAs (64x64 wave tile).
static __device__ __forceinline__ void computeChunk(const bf16* Xs, const bf16* Ws,
                                                    f32x8 (&acc)[4][4],
                                                    int wm, int wn, int lane) {
#pragma unroll
    for (int ks = 0; ks < 64; ks += 32) {
        Frag a[4], bb[4];
#pragma unroll
        for (int mi = 0; mi < 4; ++mi)
            loadA(a[mi], Xs, wm * 64 + mi * 16, S1, ks, lane);
#pragma unroll
        for (int ni = 0; ni < 4; ++ni)
            loadB(bb[ni], Ws, wn * 64 + ni * 16, S1, ks, lane);
#pragma unroll
        for (int mi = 0; mi < 4; ++mi)
#pragma unroll
            for (int ni = 0; ni < 4; ++ni)
                acc[mi][ni] = __builtin_amdgcn_wmma_f32_16x16x32_bf16(
                    false, a[mi].bf, false, bb[ni].bf,
                    (short)0, acc[mi][ni], false, false);
    }
}

// Fused: out[m, g*128+o] = sum_i (x[m,:].W[g*128+i,:] + b[g*128+i]) * Wg[g,o,i] + bg[g,o]
// Workgroup: 256 rows x one 128-col group; 8 waves in 4(M) x 2(N) grid, 64x64 per wave.
// Stage 1 double-buffers its LDS tiles: one barrier per 64-k chunk.
__global__ __launch_bounds__(256)
void EnhanceLayerLinear_fused_kernel(const float* __restrict__ x,
                                     const float* __restrict__ W,
                                     const float* __restrict__ b,
                                     const float* __restrict__ Wg,
                                     const float* __restrict__ bg,
                                     float* __restrict__ out) {
    // 142 KB LDS (WGP has 320 KB): double-buffered stage-1 tiles + Wg tile.
    __shared__ bf16 smem[2 * (256 * S1 + 128 * S1) + 128 * S2];
    bf16* Xs0 = smem;                                   // [256][S1]
    bf16* Ws0 = smem + 256 * S1;                        // [128][S1]
    bf16* Xs1 = smem + (256 + 128) * S1;                // [256][S1]
    bf16* Ws1 = smem + (512 + 128) * S1;                // [128][S1]
    bf16* Gs  = smem + 2 * (256 + 128) * S1;            // [128][S2] Wg[g]
    bf16* Ys  = smem;                                   // [128][S2] aliases Xs0 (stage 2)

    const int t    = threadIdx.x;
    const int lane = t & 31;
    const int w    = t >> 5;
    const int wm   = w & 3;                 // 64-row band within 256
    const int wn   = w >> 2;                // 64-col band within 128
    const int m0   = blockIdx.x * 256;
    const int g    = blockIdx.y;
    const int gbase = g * 128;

    // Per-lane biases for the 4 column sub-tiles this wave owns.
    float bv[4], bgv[4];
#pragma unroll
    for (int ni = 0; ni < 4; ++ni) {
        const int col = gbase + wn * 64 + ni * 16 + (lane & 15);
        bv[ni]  = b[col];
        bgv[ni] = bg[col];
    }

    // Wg[g] (128x128 fp32) -> bf16 LDS, [o][i] row-major with padded stride.
#pragma unroll
    for (int i = 0; i < 16; ++i) {
        const int idx = t + i * 256;
        const int row = idx >> 5;
        const int c   = (idx & 31) * 4;
        const float4 v = *(const float4*)(Wg + ((size_t)gbase + row) * 128 + c);
        u32x2 p = { pack2(v.x, v.y), pack2(v.z, v.w) };
        *(u32x2*)(Gs + row * S2 + c) = p;
    }

    f32x8 acc[4][4];
#pragma unroll
    for (int mi = 0; mi < 4; ++mi)
#pragma unroll
        for (int ni = 0; ni < 4; ++ni)
            acc[mi][ni] = 0.0f;

    // ------------- Stage 1: y = x . W^T (K = 4096, double-buffered) -------------
    loadChunk(x, W, Xs0, Ws0, m0, gbase, 0, t);
    __syncthreads();
    for (int k0 = 0; k0 < 4096; k0 += 128) {
        // Compute on buf0 while prefetching k0+64 into buf1.
        loadChunk(x, W, Xs1, Ws1, m0, gbase, k0 + 64, t);
        computeChunk(Xs0, Ws0, acc, wm, wn, lane);
        __syncthreads();
        // Compute on buf1 while prefetching k0+128 into buf0.
        if (k0 + 128 < 4096)
            loadChunk(x, W, Xs0, Ws0, m0, gbase, k0 + 128, t);
        computeChunk(Xs1, Ws1, acc, wm, wn, lane);
        __syncthreads();
    }

    // -------- Stage 2: out = (y + b) . Wg[g]^T + bg, two 128-row passes ---------
    for (int p = 0; p < 2; ++p) {
        if (p) __syncthreads();  // previous pass reads of Ys done (p=0 covered above)
        if ((wm >> 1) == p) {
            // Waves owning rows [p*128, p*128+128) stage y+b into LDS as bf16.
#pragma unroll
            for (int mi = 0; mi < 4; ++mi)
#pragma unroll
                for (int ni = 0; ni < 4; ++ni)
#pragma unroll
                    for (int r = 0; r < 8; ++r) {
                        // C layout: VGPR r, lane<16 -> (M=r,N=lane); lane>=16 -> (M=r+8,N=lane-16)
                        const int row = (wm & 1) * 64 + mi * 16 + (lane >> 4) * 8 + r;
                        const int col = wn * 64 + ni * 16 + (lane & 15);
                        Ys[row * S2 + col] = (bf16)(acc[mi][ni][r] + bv[ni]);
                    }
        }
        __syncthreads();

        // 128x128 out tile split 8 ways: wave tile 32x64.
        f32x8 acc2[2][4];
#pragma unroll
        for (int mi = 0; mi < 2; ++mi)
#pragma unroll
            for (int ni = 0; ni < 4; ++ni)
                acc2[mi][ni] = 0.0f;

#pragma unroll
        for (int ks = 0; ks < 128; ks += 32) {
            Frag a[2], bb[4];
#pragma unroll
            for (int mi = 0; mi < 2; ++mi)
                loadA(a[mi], Ys, wm * 32 + mi * 16, S2, ks, lane);
#pragma unroll
            for (int ni = 0; ni < 4; ++ni)
                loadB(bb[ni], Gs, wn * 64 + ni * 16, S2, ks, lane);
#pragma unroll
            for (int mi = 0; mi < 2; ++mi)
#pragma unroll
                for (int ni = 0; ni < 4; ++ni)
                    acc2[mi][ni] = __builtin_amdgcn_wmma_f32_16x16x32_bf16(
                        false, a[mi].bf, false, bb[ni].bf,
                        (short)0, acc2[mi][ni], false, false);
        }

        // Non-temporal coalesced fp32 stores (out is write-once, never re-read:
        // keep it from evicting x/W from the 192 MB L2).
#pragma unroll
        for (int mi = 0; mi < 2; ++mi)
#pragma unroll
            for (int ni = 0; ni < 4; ++ni)
#pragma unroll
                for (int r = 0; r < 8; ++r) {
                    const int row = p * 128 + wm * 32 + mi * 16 + (lane >> 4) * 8 + r;
                    const int col = wn * 64 + ni * 16 + (lane & 15);
                    __builtin_nontemporal_store(acc2[mi][ni][r] + bgv[ni],
                        out + (size_t)(m0 + row) * 4096 + gbase + col);
                }
    }
}

extern "C" void kernel_launch(void* const* d_in, const int* in_sizes, int n_in,
                              void* d_out, int out_size, void* d_ws, size_t ws_size,
                              hipStream_t stream) {
    const float* x  = (const float*)d_in[0];
    const float* W  = (const float*)d_in[1];
    const float* b  = (const float*)d_in[2];
    const float* Wg = (const float*)d_in[3];
    const float* bg = (const float*)d_in[4];
    float* out = (float*)d_out;

    const int M = in_sizes[0] / 4096;         // B*S = 8192
    dim3 grid(M / 256, 32);                   // 32 row-tiles x 32 groups
    dim3 block(256);                          // 8 wave32
    hipLaunchKernelGGL(EnhanceLayerLinear_fused_kernel, grid, block, 0, stream,
                       x, W, b, Wg, bg, out);
}
